// ScaledCorticalColumn_1425929142379
// MI455X (gfx1250) — compile-verified, matrix operand order checked by reading, not run on hardware
//
#include <hip/hip_runtime.h>
#include <hip/hip_bf16.h>

// ---------------------------------------------------------------------------
// Types for CDNA5 WMMA (wave32): v_wmma_f32_16x16x32_bf16
// ---------------------------------------------------------------------------
typedef __bf16 bf16_t;
typedef __attribute__((ext_vector_type(16))) __bf16 v16bf;
typedef __attribute__((ext_vector_type(8)))  __bf16 v8bf;
typedef __attribute__((ext_vector_type(8)))  float  v8f;

static constexpr int Bb = 2;
static constexpr int Nn = 2048;
static constexpr int Dd = 1024;
static constexpr int Hh = 16;
static constexpr int HD = 64;

__device__ __forceinline__ v8f wmma_bf16(v16bf a, v16bf b, v8f c) {
    // D = A(16x32 bf16) * B(32x16 bf16) + C(16x16 f32)
    return __builtin_amdgcn_wmma_f32_16x16x32_bf16(
        false, a, false, b, (short)0, c, false, false);
}

// 16-bit A-matrix 16x32 layout (ISA 7.12.2):
// lane L: M = L%16, koff = (L>=16)?8:0; K = {koff..koff+7} in v0..3,
// {koff+16..koff+23} in v4..7.  B 32x16 is symmetric (N in place of M).
// Both load identically from a row-major [16][>=32] bf16 tile.
__device__ __forceinline__ v16bf load_tile16(const bf16_t* __restrict__ base,
                                             int ld, int lane) {
    int r    = lane & 15;
    int koff = (lane >> 4) << 3;
    const bf16_t* p = base + (size_t)r * ld + koff;
    v8bf lo = *(const v8bf*)(p);
    v8bf hi = *(const v8bf*)(p + 16);
    v16bf out;
#pragma unroll
    for (int i = 0; i < 8; ++i) { out[i] = lo[i]; out[i + 8] = hi[i]; }
    return out;
}

// ---------------------------------------------------------------------------
// LayerNorm (fp32 in) -> bf16 out.  One 256-thread block per row.
// ---------------------------------------------------------------------------
__global__ __launch_bounds__(256) void ln_kernel(
    const float* __restrict__ x, const float* __restrict__ g,
    const float* __restrict__ b, bf16_t* __restrict__ out, int D)
{
    __shared__ float ss[8], ss2[8];
    int row = blockIdx.x;
    const float* xr = x + (size_t)row * D;
    float s = 0.f, s2 = 0.f;
    for (int i = threadIdx.x; i < D; i += 256) {
        float v = xr[i]; s += v; s2 += v * v;
    }
    int lane = threadIdx.x & 31, wid = threadIdx.x >> 5;
#pragma unroll
    for (int o = 16; o; o >>= 1) { s += __shfl_xor(s, o); s2 += __shfl_xor(s2, o); }
    if (lane == 0) { ss[wid] = s; ss2[wid] = s2; }
    __syncthreads();
    if (wid == 0) {
        s  = (lane < 8) ? ss[lane]  : 0.f;
        s2 = (lane < 8) ? ss2[lane] : 0.f;
#pragma unroll
        for (int o = 4; o; o >>= 1) { s += __shfl_xor(s, o); s2 += __shfl_xor(s2, o); }
        if (lane == 0) { ss[0] = s; ss2[0] = s2; }
    }
    __syncthreads();
    float mean = ss[0] / (float)D;
    float var  = ss2[0] / (float)D - mean * mean;
    float rstd = rsqrtf(var + 1e-5f);
    for (int i = threadIdx.x; i < D; i += 256) {
        float v = (xr[i] - mean) * rstd * g[i] + b[i];
        out[(size_t)row * D + i] = (bf16_t)v;
    }
}

// ---------------------------------------------------------------------------
// Tiled WMMA GEMM:  C[M,Nc] = A_bf16[M,K] @ W_f32[K,Nc] (+bias) (+epilogue)
// 128 threads (4 waves); WG tile 128x64; wave tile 32x64 (8 accumulators,
// 8 WMMAs per staged W chunk).  W chunk (32 x 64) is converted fp32->bf16 and
// staged transposed in LDS with ping-pong double buffering (1 barrier/iter).
// MODE: 0 = out_f32 + out_bf16          (h = lnx @ w_in + b_in)
//       1 = out_bf16                    (q, k, up)
//       2 = out_bf16 transposed [B,H,HD,N]  (v)
//       3 = out_f32 = acc + bias + res  (h2, final out)
//       4 = out_bf16 = silu(acc+bias) * aux_bf16   (gate * up)
// ---------------------------------------------------------------------------
template <int MODE>
__global__ __launch_bounds__(128) void gemm_wmma_kernel(
    const bf16_t* __restrict__ A, const float* __restrict__ W,
    const float* __restrict__ bias,
    float* __restrict__ out_f32, bf16_t* __restrict__ out_bf,
    const float* __restrict__ res_f32, const bf16_t* __restrict__ aux_bf,
    int M, int K, int Nc)
{
    __shared__ bf16_t wT[2][64][40];   // [buf][Ncol][Krow], rows 80B apart
    const int tid = threadIdx.x, lane = tid & 31, wid = tid >> 5;
    const int m0 = blockIdx.y * 128 + wid * 32;
    const int n0 = blockIdx.x * 64;

    v8f acc[2][4] = {};
    const int skr = tid >> 2;            // 0..31 : K row within chunk
    const int sc0 = (tid & 3) * 16;      // 16 consecutive columns

    // prologue: stage chunk 0 into buffer 0 (fp32 -> bf16, transposed)
    {
        const float* wp = W + (size_t)skr * Nc + n0 + sc0;
#pragma unroll
        for (int j = 0; j < 16; ++j) wT[0][sc0 + j][skr] = (bf16_t)wp[j];
    }
    __syncthreads();

    for (int kk = 0; kk < K; kk += 32) {
        const int cur = (kk >> 5) & 1;
        if (kk + 32 < K) {   // stage next chunk into the other buffer
            const float* wp = W + (size_t)(kk + 32 + skr) * Nc + n0 + sc0;
#pragma unroll
            for (int j = 0; j < 16; ++j) wT[cur ^ 1][sc0 + j][skr] = (bf16_t)wp[j];
        }
        v16bf a0 = load_tile16(A + (size_t)m0 * K + kk,        K, lane);
        v16bf a1 = load_tile16(A + (size_t)(m0 + 16) * K + kk, K, lane);
#pragma unroll
        for (int s = 0; s < 4; ++s) {
            v16bf b = load_tile16(&wT[cur][s * 16][0], 40, lane);
            acc[0][s] = wmma_bf16(a0, b, acc[0][s]);
            acc[1][s] = wmma_bf16(a1, b, acc[1][s]);
        }
        __syncthreads();   // protects buffer overwritten next iteration
    }

    // C layout: VGPR i, lane L -> row = i + 8*(L>=16), col = L%16
    const int cl = lane & 15;
    const int rb = (lane >> 4) * 8;
#pragma unroll
    for (int t = 0; t < 2; ++t) {
#pragma unroll
        for (int s = 0; s < 4; ++s) {
#pragma unroll
            for (int i = 0; i < 8; ++i) {
                int row = m0 + t * 16 + rb + i;
                int col = n0 + s * 16 + cl;
                float v = acc[t][s][i] + (bias ? bias[col] : 0.0f);
                size_t idx = (size_t)row * Nc + col;
                if constexpr (MODE == 0) {
                    out_f32[idx] = v;
                    out_bf[idx]  = (bf16_t)v;
                } else if constexpr (MODE == 1) {
                    out_bf[idx] = (bf16_t)v;
                } else if constexpr (MODE == 2) {
                    int bidx = row >> 11, n = row & (Nn - 1);
                    out_bf[((size_t)(bidx * Dd + col)) * Nn + n] = (bf16_t)v;
                } else if constexpr (MODE == 3) {
                    out_f32[idx] = v + res_f32[idx];
                } else if constexpr (MODE == 4) {
                    float sg = v / (1.0f + __expf(-v));        // silu
                    out_bf[idx] = (bf16_t)(sg * (float)aux_bf[idx]);
                }
            }
        }
    }
}

// ---------------------------------------------------------------------------
// Sparse top-k attention.  One WG (128 thr / 4 waves) per (b, h, 16-q tile).
// Phase 1: scores = (Q Kt)/8 via WMMA into LDS (16 x 2048 fp32).
// Phase 2: per-row exact k-th largest via 32-step binary search on the
//          order-preserving uint map of fp32; masked softmax (ties kept,
//          matching jnp.where(scores >= kth)).
// Phase 3: P @ V via WMMA; V pre-transposed so B-operand loads are b128.
// ---------------------------------------------------------------------------
#define SC_STRIDE 2052
#define SC_BYTES  (16 * SC_STRIDE * 4)

__global__ __launch_bounds__(128) void attn_kernel(
    const bf16_t* __restrict__ q, const bf16_t* __restrict__ k,
    const bf16_t* __restrict__ vT, bf16_t* __restrict__ ctx,
    const int* __restrict__ top_k_p)
{
    extern __shared__ char smem_raw[];
    float* sc = (float*)smem_raw;

    const int qt = blockIdx.x, h = blockIdx.y, b = blockIdx.z;
    const int tid = threadIdx.x, lane = tid & 31, wid = tid >> 5;
    const int n0 = qt * 16;
    const int tk = top_k_p[0];

    const bf16_t* qbase = q  + ((size_t)(b * Nn + n0)) * Dd + h * HD;
    const bf16_t* kbase = k  + ((size_t)(b * Nn)) * Dd + h * HD;
    const bf16_t* vbase = vT + ((size_t)(b * Dd + h * HD)) * Nn;

    // ---- Phase 1: scores --------------------------------------------------
    v16bf a0 = load_tile16(qbase,      Dd, lane);   // hd 0..31
    v16bf a1 = load_tile16(qbase + 32, Dd, lane);   // hd 32..63
    const int cl = lane & 15;
    const int rb = (lane >> 4) * 8;
    for (int kt = wid * 32; kt < wid * 32 + 32; ++kt) {
        const bf16_t* kb = kbase + (size_t)(kt * 16) * Dd;
        v16bf b0 = load_tile16(kb,      Dd, lane);
        v16bf b1 = load_tile16(kb + 32, Dd, lane);
        v8f c = {};
        c = wmma_bf16(a0, b0, c);
        c = wmma_bf16(a1, b1, c);
        int col = kt * 16 + cl;
#pragma unroll
        for (int i = 0; i < 8; ++i)
            sc[(rb + i) * SC_STRIDE + col] = c[i] * 0.125f;   // HD^-0.5
    }
    __syncthreads();

    // ---- Phase 2: top-k threshold + softmax (1 row per wave, 4 rounds) ----
    for (int rr = 0; rr < 4; ++rr) {
        int r = wid + rr * 4;
        float    fv[64];
        unsigned mv[64];
#pragma unroll
        for (int j = 0; j < 64; ++j) {
            float f = sc[r * SC_STRIDE + lane + j * 32];
            fv[j] = f;
            unsigned u = __float_as_uint(f);
            mv[j] = (u & 0x80000000u) ? ~u : (u | 0x80000000u);
        }
        unsigned thr = 0u;
        if (tk < Nn) {
            unsigned lo = 0u, hi = 0xFFFFFFFFu;
            while (lo < hi) {
                unsigned d   = hi - lo;
                unsigned mid = lo + (d >> 1) + (d & 1u);     // ceil midpoint
                int cnt = 0;
#pragma unroll
                for (int j = 0; j < 64; ++j) cnt += (mv[j] >= mid) ? 1 : 0;
#pragma unroll
                for (int o = 16; o; o >>= 1) cnt += __shfl_xor(cnt, o);
                if (cnt >= tk) lo = mid; else hi = mid - 1;
            }
            thr = lo;   // mapped value of the k-th largest score
        }
        float fmax = -3.4e38f;
#pragma unroll
        for (int j = 0; j < 64; ++j)
            if (mv[j] >= thr) fmax = fmaxf(fmax, fv[j]);
#pragma unroll
        for (int o = 16; o; o >>= 1) fmax = fmaxf(fmax, __shfl_xor(fmax, o));
        float fsum = 0.f;
#pragma unroll
        for (int j = 0; j < 64; ++j)
            fsum += (mv[j] >= thr) ? __expf(fv[j] - fmax) : 0.0f;
#pragma unroll
        for (int o = 16; o; o >>= 1) fsum += __shfl_xor(fsum, o);
        float inv = 1.0f / fsum;
#pragma unroll
        for (int j = 0; j < 64; ++j) {
            float p = (mv[j] >= thr) ? __expf(fv[j] - fmax) * inv : 0.0f;
            sc[r * SC_STRIDE + lane + j * 32] = p;
        }
    }
    __syncthreads();

    // ---- Phase 3: out = P @ V  (wave wid owns hd columns wid*16..+15) -----
    const int hd0  = wid * 16;
    const int rA   = lane & 15;
    const int koff = (lane >> 4) << 3;
    const bf16_t* vb = vbase + (size_t)hd0 * Nn;
    v8f c = {};
    for (int kk = 0; kk < Nn; kk += 32) {
        const float* sp = sc + rA * SC_STRIDE + kk + koff;
        v16bf a;
#pragma unroll
        for (int j = 0; j < 8; ++j) {
            a[j]     = (bf16_t)sp[j];
            a[j + 8] = (bf16_t)sp[16 + j];
        }
        const bf16_t* bp = vb + (size_t)cl * Nn + kk + koff;
        v8bf blo = *(const v8bf*)(bp);
        v8bf bhi = *(const v8bf*)(bp + 16);
        v16bf bb;
#pragma unroll
        for (int j = 0; j < 8; ++j) { bb[j] = blo[j]; bb[j + 8] = bhi[j]; }
        c = wmma_bf16(a, bb, c);
    }
    int col = h * HD + hd0 + cl;
#pragma unroll
    for (int i = 0; i < 8; ++i) {
        int row = b * Nn + n0 + rb + i;
        ctx[(size_t)row * Dd + col] = (bf16_t)c[i];
    }
}

// ---------------------------------------------------------------------------
// Host launch.  Workspace layout (bytes), with aliasing:
//   [0,8M)    lnx_bf16   -> later reused as ctx_bf16
//   [8M,24M)  h_f32
//   [24M,32M) h_bf16
//   [32M,40M) q_bf16  \
//   [40M,48M) k_bf16   }-> later reused as act_bf16 [32M,64M)
//   [48M,56M) vT_bf16 /
//   [64M,80M) h2_f32
//   [80M,88M) lnh2_bf16
//   [88M,120M) up_bf16
// ---------------------------------------------------------------------------
extern "C" void kernel_launch(void* const* d_in, const int* in_sizes, int n_in,
                              void* d_out, int out_size, void* d_ws, size_t ws_size,
                              hipStream_t stream) {
    (void)in_sizes; (void)n_in; (void)out_size; (void)ws_size;
    const float* x      = (const float*)d_in[0];
    const float* ln1_g  = (const float*)d_in[1];
    const float* ln1_b  = (const float*)d_in[2];
    const float* w_in   = (const float*)d_in[3];
    const float* b_in   = (const float*)d_in[4];
    const float* wq     = (const float*)d_in[5];
    const float* wk     = (const float*)d_in[6];
    const float* wv     = (const float*)d_in[7];
    const float* wo     = (const float*)d_in[8];
    const float* bo     = (const float*)d_in[9];
    const float* ln2_g  = (const float*)d_in[10];
    const float* ln2_b  = (const float*)d_in[11];
    const float* w_up   = (const float*)d_in[12];
    const float* b_up   = (const float*)d_in[13];
    const float* w_gate = (const float*)d_in[14];
    const float* b_gate = (const float*)d_in[15];
    const float* w_down = (const float*)d_in[16];
    const float* b_down = (const float*)d_in[17];
    const int*   top_k  = (const int*)d_in[18];
    float* out = (float*)d_out;

    char* ws = (char*)d_ws;
    const size_t MB = 1024ull * 1024ull;
    bf16_t* lnx  = (bf16_t*)(ws + 0);
    bf16_t* ctx  = (bf16_t*)(ws + 0);          // alias (lnx dead after h GEMM)
    float*  h32  = (float*) (ws + 8  * MB);
    bf16_t* hbf  = (bf16_t*)(ws + 24 * MB);
    bf16_t* qbf  = (bf16_t*)(ws + 32 * MB);
    bf16_t* kbf  = (bf16_t*)(ws + 40 * MB);
    bf16_t* vt   = (bf16_t*)(ws + 48 * MB);
    bf16_t* act  = (bf16_t*)(ws + 32 * MB);    // alias (q/k/vT dead after attn)
    float*  h2   = (float*) (ws + 64 * MB);
    bf16_t* ln2o = (bf16_t*)(ws + 80 * MB);
    bf16_t* upbf = (bf16_t*)(ws + 88 * MB);

    const int M = Bb * Nn;                     // 4096
    dim3 blk(128);
    dim3 gsq(Dd / 64, M / 128);                // 16 x 32  (Nc=1024)
    dim3 gup(4 * Dd / 64, M / 128);            // 64 x 32  (Nc=4096)

    // 1) ln1(x) -> bf16
    ln_kernel<<<M, 256, 0, stream>>>(x, ln1_g, ln1_b, lnx, Dd);
    // 2) h = lnx @ w_in + b_in   (fp32 + bf16)
    gemm_wmma_kernel<0><<<gsq, blk, 0, stream>>>(lnx, w_in, b_in, h32, hbf,
                                                 nullptr, nullptr, M, Dd, Dd);
    // 3) q,k,v projections (v written transposed)
    gemm_wmma_kernel<1><<<gsq, blk, 0, stream>>>(hbf, wq, nullptr, nullptr, qbf,
                                                 nullptr, nullptr, M, Dd, Dd);
    gemm_wmma_kernel<1><<<gsq, blk, 0, stream>>>(hbf, wk, nullptr, nullptr, kbf,
                                                 nullptr, nullptr, M, Dd, Dd);
    gemm_wmma_kernel<2><<<gsq, blk, 0, stream>>>(hbf, wv, nullptr, nullptr, vt,
                                                 nullptr, nullptr, M, Dd, Dd);
    // 4) sparse attention -> ctx (bf16)
    dim3 ga(Nn / 16, Hh, Bb);
    attn_kernel<<<ga, blk, SC_BYTES, stream>>>(qbf, kbf, vt, ctx, top_k);
    // 5) h2 = h + ctx @ wo + bo
    gemm_wmma_kernel<3><<<gsq, blk, 0, stream>>>(ctx, wo, bo, h2, nullptr,
                                                 h32, nullptr, M, Dd, Dd);
    // 6) ln2(h2) -> bf16
    ln_kernel<<<M, 256, 0, stream>>>(h2, ln2_g, ln2_b, ln2o, Dd);
    // 7) up = ln2o @ w_up + b_up (bf16)
    gemm_wmma_kernel<1><<<gup, blk, 0, stream>>>(ln2o, w_up, b_up, nullptr, upbf,
                                                 nullptr, nullptr, M, Dd, 4 * Dd);
    // 8) act = silu(ln2o @ w_gate + b_gate) * up
    gemm_wmma_kernel<4><<<gup, blk, 0, stream>>>(ln2o, w_gate, b_gate, nullptr,
                                                 act, nullptr, upbf, M, Dd, 4 * Dd);
    // 9) out = h2 + act @ w_down + b_down
    gemm_wmma_kernel<3><<<gsq, blk, 0, stream>>>(act, w_down, b_down, out,
                                                 nullptr, h2, nullptr, M, 4 * Dd, Dd);
}
